// Codebook_66211215835652
// MI455X (gfx1250) — matrix operand order repaired and checked by reading, not compiled, not run on hardware
//
#include <hip/hip_runtime.h>
#include <hip/hip_bf16.h>

// ---------------- problem constants ----------------
#define KCODES   512
#define DIM      256
#define NPIX     65536          // 64*32*32
#define MTILE    64             // pixels per workgroup
#define NBLK     (NPIX / MTILE) // 1024
#define XSTRIDE  262144         // DIM*1024 floats per batch image
#define LOSS_SCALE (1.25f / 16777216.0f)  // (1+BETA)/ (b*d*h*w)

// ---------------- LDS layout (dwords) ----------------
// padded rows: 128 data dwords + 4 pad = 132 dwords (528B, 16B aligned,
// row-to-row bank stride of 4 -> conflict-free 16-lane b128 fragment loads)
#define LROW     132
#define OFF_LUT  0                         // 512*132 = 67584
#define OFF_X    67584                     // 64*132  = 8448
#define OFF_C2   (OFF_X + 8448)            // 512
#define OFF_BD   (OFF_C2 + 512)            // 128 (2 halves * 64 rows)
#define OFF_BI   (OFF_BD + 128)            // 128
#define OFF_Q    (OFF_BI + 128)            // 64
#define SMEM_DW  (OFF_Q + 64)              // 76864 dwords = 307456 B <= 320KB
// x_e f32 staging aliases the (dead) LUT region: 64 rows * 260 dwords
#define XEROW    260

typedef __attribute__((ext_vector_type(16))) __bf16 v16bf;
typedef __attribute__((ext_vector_type(8)))  __bf16 v8bf;
typedef __attribute__((ext_vector_type(8)))  float  v8f;

static __device__ __forceinline__ unsigned pack_bf16(float a, float b) {
  __bf16 ha = (__bf16)a, hb = (__bf16)b;
  unsigned short ua = __builtin_bit_cast(unsigned short, ha);
  unsigned short ub = __builtin_bit_cast(unsigned short, hb);
  return (unsigned)ua | ((unsigned)ub << 16);
}

// ------------------------------------------------------------------
// Kernel 1: LUT f32 -> packed bf16 in ws, plus |c_k|^2 per code.
// grid = 512 blocks, 128 threads
// ------------------------------------------------------------------
__global__ void vq_prep_kernel(const float* __restrict__ lut,
                               unsigned* __restrict__ wlut,
                               float* __restrict__ wc2) {
  const int k = blockIdx.x;
  const int t = threadIdx.x;           // 0..127
  float f0 = lut[k * DIM + 2 * t];
  float f1 = lut[k * DIM + 2 * t + 1];
  wlut[k * (DIM / 2) + t] = pack_bf16(f0, f1);
  float s = f0 * f0 + f1 * f1;
#pragma unroll
  for (int off = 16; off; off >>= 1) s += __shfl_xor(s, off, 32);
  __shared__ float red[4];
  if ((t & 31) == 0) red[t >> 5] = s;
  __syncthreads();
  if (t == 0) wc2[k] = red[0] + red[1] + red[2] + red[3];
}

// ------------------------------------------------------------------
// Kernel 2: main VQ kernel. grid = 1024 blocks, 256 threads (8 waves)
// ------------------------------------------------------------------
__global__ void __launch_bounds__(256)
vq_main_kernel(const float* __restrict__ x,
               const float* __restrict__ lutf32,
               const unsigned* __restrict__ wlut,
               const float* __restrict__ wc2,
               float* __restrict__ out_xe,
               float* __restrict__ out_q,
               float* __restrict__ partials) {
  __shared__ unsigned smem[SMEM_DW];

  const int t    = threadIdx.x;
  const int lane = t & 31;
  const int w    = t >> 5;        // wave id 0..7
  const int lm   = lane & 15;     // column / row-in-block selector
  const int half = lane >> 4;     // lane half (K/M split per ISA layout)

  const int n0 = blockIdx.x * MTILE;      // first pixel of this tile
  const int b  = n0 >> 10;                // batch image
  const int p0 = n0 & 1023;               // pixel offset inside image
  const float* xg = x + (size_t)b * XSTRIDE + p0;

  // ---- stage codebook (bf16, padded rows) : 16384 uint4 total ----
  {
    const uint4* src = (const uint4*)wlut;
#pragma unroll 4
    for (int i = 0; i < 64; ++i) {
      int idx = i * 256 + t;              // uint4 index
      int row = idx >> 5;                 // 32 uint4 per code row
      int col = idx & 31;
      *(uint4*)&smem[OFF_LUT + row * LROW + col * 4] = src[idx];
    }
  }
  // ---- stage x tile (transpose [d][p] -> [p][d], f32 -> packed bf16) ----
  {
    const int p = t & 63;
    const int g = t >> 6;                 // 0..3
#pragma unroll 4
    for (int i = 0; i < 32; ++i) {
      int d = i * 8 + g * 2;
      float f0 = xg[(size_t)d * 1024 + p];
      float f1 = xg[(size_t)(d + 1) * 1024 + p];
      smem[OFF_X + p * LROW + (d >> 1)] = pack_bf16(f0, f1);
    }
  }
  // ---- stage c2 ----
  {
    float* s_c2 = (float*)&smem[OFF_C2];
    s_c2[t]       = wc2[t];
    s_c2[t + 256] = wc2[t + 256];
  }
  __syncthreads();

  // ---- compute phase: wave w -> row block (w&3), code half (w>>2) ----
  const int rb = w & 3;        // 16-row block within the 64-pixel tile
  const int ch = w >> 2;       // code half: 0 -> codes 0..255, 1 -> 256..511
  const float* s_c2 = (const float*)&smem[OFF_C2];

  // Preload A fragments: 8 ktiles of 16x32 bf16 (ISA 7.12.2 layout)
  v16bf afrag[8];
  {
    const int abase = (rb * 16 + lm) * LROW + half * 4;
#pragma unroll
    for (int kt = 0; kt < 8; ++kt) {
      int o = OFF_X + abase + kt * 16;
      v8bf lo = *(const v8bf*)&smem[o];
      v8bf hi = *(const v8bf*)&smem[o + 8];
      afrag[kt] = __builtin_shufflevector(lo, hi, 0, 1, 2, 3, 4, 5, 6, 7,
                                          8, 9, 10, 11, 12, 13, 14, 15);
    }
  }

  float best[8];
  int   bidx[8];
#pragma unroll
  for (int v = 0; v < 8; ++v) { best[v] = 3.0e38f; bidx[v] = 0; }

  for (int ct = 0; ct < 16; ++ct) {           // 16 code tiles per wave
    const int ctbase = (ch * 16 + ct) * 16;   // first code of tile
    const int code   = ctbase + lm;           // this lane's code column
    // B fragments: column = code row, K in ISA table order (matches A)
    v16bf bfrag[8];
    {
      const int bbase = code * LROW + half * 4;
#pragma unroll
      for (int kt = 0; kt < 8; ++kt) {
        int o = OFF_LUT + bbase + kt * 16;
        v8bf lo = *(const v8bf*)&smem[o];
        v8bf hi = *(const v8bf*)&smem[o + 8];
        bfrag[kt] = __builtin_shufflevector(lo, hi, 0, 1, 2, 3, 4, 5, 6, 7,
                                            8, 9, 10, 11, 12, 13, 14, 15);
      }
    }
    v8f acc = {};
#pragma unroll
    for (int kt = 0; kt < 8; ++kt) {
      acc = __builtin_amdgcn_wmma_f32_16x16x32_bf16(
          false, afrag[kt], false, bfrag[kt], (short)0, acc, false, false);
    }
    const float c2v = s_c2[code];
#pragma unroll
    for (int v = 0; v < 8; ++v) {
      float dist = c2v - 2.0f * acc[v];       // |c|^2 - 2 x.c (argmin-equiv)
      if (dist < best[v]) { best[v] = dist; bidx[v] = code; }
    }
  }

  // ---- reduce over the 16 columns held in each lane half ----
#pragma unroll
  for (int v = 0; v < 8; ++v) {
    float bd = best[v]; int bi = bidx[v];
#pragma unroll
    for (int off = 1; off < 16; off <<= 1) {
      float od = __shfl_xor(bd, off, 32);
      int   oi = __shfl_xor(bi, off, 32);
      if (od < bd || (od == bd && oi < bi)) { bd = od; bi = oi; }
    }
    best[v] = bd; bidx[v] = bi;
  }
  // lanes 0 / 16 hold the per-row winners (rows v and v+8 of the block)
  {
    float* s_bd = (float*)&smem[OFF_BD];
    int*   s_bi = (int*)&smem[OFF_BI];
    if (lm == 0) {
      const int rbase = rb * 16 + half * 8;
#pragma unroll
      for (int v = 0; v < 8; ++v) {
        s_bd[ch * 64 + rbase + v] = best[v];
        s_bi[ch * 64 + rbase + v] = bidx[v];
      }
    }
  }
  __syncthreads();

  // ---- merge the two code halves, emit q ----
  {
    float* s_bd = (float*)&smem[OFF_BD];
    int*   s_bi = (int*)&smem[OFF_BI];
    int*   s_q  = (int*)&smem[OFF_Q];
    if (t < MTILE) {
      float d0 = s_bd[t], d1 = s_bd[64 + t];
      int   i0 = s_bi[t], i1 = s_bi[64 + t];
      int q = (d1 < d0) ? i1 : i0;          // tie -> lower index (half 0)
      s_q[t] = q;
      out_q[n0 + t] = (float)q;
    }
  }
  __syncthreads();

  // ---- stage x_e rows (f32) into LDS, aliasing the dead LUT region ----
  {
    float* s_xe = (float*)&smem[OFF_LUT];
    const int* s_q = (const int*)&smem[OFF_Q];
    const uint4* lut4 = (const uint4*)lutf32;
    const int r = t >> 2, seg = t & 3;
    const int q = s_q[r];
#pragma unroll 4
    for (int i = 0; i < 16; ++i) {
      int col4 = seg * 16 + i;              // float4 column 0..63
      *(uint4*)&s_xe[r * XEROW + col4 * 4] =
          lut4[(size_t)q * (DIM / 4) + col4];
    }
  }
  __syncthreads();

  // ---- write x_e (coalesced over pixels) + accumulate squared diff ----
  float lsum = 0.0f;
  {
    const float* s_xe = (const float*)&smem[OFF_LUT];
    float* og = out_xe + (size_t)b * XSTRIDE + p0;
    const int p = t & 63, dg = t >> 6;
#pragma unroll 4
    for (int i = 0; i < 64; ++i) {
      int d = dg * 64 + i;
      float xe = s_xe[p * XEROW + d];
      float xv = xg[(size_t)d * 1024 + p];
      float df = xv - xe;
      lsum += df * df;
      og[(size_t)d * 1024 + p] = xe;
    }
  }
#pragma unroll
  for (int off = 16; off; off >>= 1) lsum += __shfl_xor(lsum, off, 32);
  {
    float* s_red = (float*)&smem[OFF_BD];   // dead after merge
    if (lane == 0) s_red[w] = lsum;
    __syncthreads();
    if (t == 0) {
      float tot = 0.0f;
#pragma unroll
      for (int i = 0; i < 8; ++i) tot += s_red[i];
      partials[blockIdx.x] = tot;
    }
  }
}

// ------------------------------------------------------------------
// Kernel 3: deterministic loss reduction. 1 block, 256 threads
// ------------------------------------------------------------------
__global__ void vq_loss_kernel(const float* __restrict__ partials,
                               float* __restrict__ out_loss) {
  const int t = threadIdx.x;
  float s = 0.0f;
  for (int i = t; i < NBLK; i += 256) s += partials[i];
#pragma unroll
  for (int off = 16; off; off >>= 1) s += __shfl_xor(s, off, 32);
  __shared__ float red[8];
  if ((t & 31) == 0) red[t >> 5] = s;
  __syncthreads();
  if (t == 0) {
    float tot = 0.0f;
#pragma unroll
    for (int i = 0; i < 8; ++i) tot += red[i];
    out_loss[0] = tot * LOSS_SCALE;
  }
}

// ------------------------------------------------------------------
extern "C" void kernel_launch(void* const* d_in, const int* in_sizes, int n_in,
                              void* d_out, int out_size, void* d_ws, size_t ws_size,
                              hipStream_t stream) {
  (void)in_sizes; (void)n_in; (void)out_size; (void)ws_size;
  const float* x   = (const float*)d_in[0];   // [64,256,32,32]
  const float* lut = (const float*)d_in[1];   // [512,256]
  float* out = (float*)d_out;                 // x_e | q_x | loss

  // workspace layout: bf16 LUT (256KB) | c2 (2KB) | partials (4KB)
  char* ws = (char*)d_ws;
  unsigned* wlut   = (unsigned*)ws;                       // 512*128 u32
  float*    wc2    = (float*)(ws + KCODES * DIM * 2);     // 512 f32
  float*    wparts = (float*)(ws + KCODES * DIM * 2 + KCODES * 4); // 1024 f32

  float* out_xe   = out;                         // 16777216
  float* out_q    = out + (size_t)NPIX * DIM;    // 65536
  float* out_loss = out_q + NPIX;                // 1

  vq_prep_kernel<<<KCODES, 128, 0, stream>>>(lut, wlut, wc2);
  vq_main_kernel<<<NBLK, 256, 0, stream>>>(x, lut, wlut, wc2,
                                           out_xe, out_q, wparts);
  vq_loss_kernel<<<1, 256, 0, stream>>>(wparts, out_loss);
}
